// MinGRULayer_13185549599203
// MI455X (gfx1250) — matrix operand order repaired and checked by reading, not compiled, not run on hardware
//
#include <hip/hip_runtime.h>
#include <hip/hip_bf16.h>
#include <math.h>

// Problem constants (match reference)
#define BB 4
#define TT 2048
#define II 512
#define SS 512
#define PP 8
#define NN (BB * TT)        // 8192 rows
#define KK (II * PP)        // 4096 dense-K
#define KSLABS (KK / 32)    // 128
#define CTILES (SS / 16)    // 32
#define CHUNKS 64
#define CLEN   (TT / CHUNKS) // 32

typedef __attribute__((ext_vector_type(16))) _Float16 v16h;
typedef __attribute__((ext_vector_type(8)))  float    v8f;
typedef int v4i32 __attribute__((vector_size(16)));   // matches builtin param type

// gfx1250 async global->LDS path (guarded: falls back to register path if the
// toolchain doesn't declare the builtins).
#if defined(__gfx1250__) && \
    __has_builtin(__builtin_amdgcn_global_load_async_to_lds_b128) && \
    __has_builtin(__builtin_amdgcn_s_wait_asynccnt)
#define USE_ASYNC_LDS 1
#else
#define USE_ASYNC_LDS 0
#endif

// ---------------------------------------------------------------------------
// Pack f32 weights [I,P,S] -> f16 in the exact WMMA B-operand register layout:
// flat index = ((ct*KSLABS + ks)*32 + lane)*16 + e
//   lane<16 : column = ct*16+lane,     K = ks*32 + (e&7) + 16*(e>>3)      (e = 0..15)
//   lane>=16: column = ct*16+lane-16,  K = ks*32 + 8 + (e&7) + 16*(e>>3)
// w flat index = K*S + col  (since [I,P,S] with K = i*P+p)
// ---------------------------------------------------------------------------
__global__ __launch_bounds__(256) void pack_b_kernel(const float* __restrict__ w,
                                                     _Float16* __restrict__ bp) {
    const int tid = blockIdx.x * 256 + threadIdx.x;     // 0 .. 2,097,151
    const int e    = tid & 15;
    const int lane = (tid >> 4) & 31;
    const int ks   = (tid >> 9) & (KSLABS - 1);
    const int ct   = tid >> 16;
    const int col  = ct * 16 + (lane & 15);
    const int K    = ks * 32 + ((lane >> 4) << 3) + (e & 7) + ((e >> 3) << 4);
    bp[tid] = (_Float16)w[(size_t)K * SS + col];
}

// ---------------------------------------------------------------------------
// Build 8 f16 piecewise-linear coefficients for one feature into a[base..base+7].
// searchsorted-left (count of knots < x), clip to [1, P-1], lerp coefficients,
// linear extrapolation at both ends. Branch-free (EXEC stays all-ones for WMMA).
// ---------------------------------------------------------------------------
__device__ __forceinline__ void build_coeff(v16h& a, int base, float x,
                                            const float* __restrict__ pp) {
    const float4 q0 = *(const float4*)(pp);
    const float4 q1 = *(const float4*)(pp + 4);
    const float p[PP] = {q0.x, q0.y, q0.z, q0.w, q1.x, q1.y, q1.z, q1.w};
    int idx = 0;
    #pragma unroll
    for (int k = 0; k < PP; ++k) idx += (p[k] < x) ? 1 : 0;
    idx = idx < 1 ? 1 : (idx > PP - 1 ? PP - 1 : idx);
    float xl = 0.f, xr = 0.f;
    #pragma unroll
    for (int k = 0; k < PP; ++k) {
        xl = (k == idx - 1) ? p[k] : xl;
        xr = (k == idx)     ? p[k] : xr;
    }
    const float t = (x - xl) / (xr - xl);
    #pragma unroll
    for (int k = 0; k < PP; ++k) {
        float c = (k == idx - 1) ? (1.0f - t) : ((k == idx) ? t : 0.0f);
        a[base + k] = (_Float16)c;
    }
}

#if USE_ASYNC_LDS
// Wave `wave` async-copies its 1KB B tile (tile ct=wave) of slab `ksn` into
// LDS buffer `qn`: 2 x GLOBAL_LOAD_ASYNC_TO_LDS_B128 per lane (ASYNCcnt).
__device__ __forceinline__ void prefetch_slab(const _Float16* __restrict__ Bp,
                                              _Float16 (&sbuf)[2][8][32][16],
                                              int colbase, int wave, int lane,
                                              int ksn, int qn) {
    const _Float16* gp = Bp +
        (((size_t)(colbase + wave) * KSLABS + ksn) * 32 + lane) * 16;
    _Float16* lp = &sbuf[qn][wave][lane][0];
    __builtin_amdgcn_global_load_async_to_lds_b128(
        (__attribute__((address_space(1))) v4i32*)(uintptr_t)gp,
        (__attribute__((address_space(3))) v4i32*)lp, 0, 0);
    __builtin_amdgcn_global_load_async_to_lds_b128(
        (__attribute__((address_space(1))) v4i32*)(uintptr_t)(gp + 8),
        (__attribute__((address_space(3))) v4i32*)(lp + 8), 0, 0);
}
#endif

// ---------------------------------------------------------------------------
// Piecewise-linear layer as WMMA GEMM.
// Block = 256 threads (8 waves), tile = 128 rows x 128 cols.
// Wave w owns rows blockIdx.x*128 + 16w .. +15, all 8 col-subtiles.
// Async path: double-buffered LDS staging of B slabs (8KB each), prefetched
// with GLOBAL_LOAD_ASYNC_TO_LDS_B128 one slab ahead; consumption via ds_load.
// ---------------------------------------------------------------------------
__global__ __launch_bounds__(256) void pl_wmma_gemm(const float* __restrict__ X,    // [N, I]
                                                    const float* __restrict__ pts,  // [I, P]
                                                    const _Float16* __restrict__ Bp,// packed
                                                    float* __restrict__ out) {      // [N, S]
    const int lane    = threadIdx.x & 31;
    const int wave    = threadIdx.x >> 5;
    const int halfsel = lane >> 4;                 // 0: lanes 0-15, 1: lanes 16-31
    const int row0    = blockIdx.x * 128 + wave * 16;
    const int r       = row0 + (lane & 15);        // A-matrix row for this lane
    const int colbase = blockIdx.y * 8;            // first col-subtile index

    v8f acc[8] = {};                               // 64 f32 accum VGPRs

#if USE_ASYNC_LDS
    __shared__ __attribute__((aligned(32))) _Float16 sbuf[2][8][32][16]; // 16 KB
    prefetch_slab(Bp, sbuf, colbase, wave, lane, 0, 0);                  // slab 0

    for (int ks = 0; ks < KSLABS; ++ks) {
        const int q   = ks & 1;
        const int ksn = (ks + 1) & (KSLABS - 1);   // wrap: last prefetch harmless
        // ---- kick off async copy of next slab into the other buffer
        prefetch_slab(Bp, sbuf, colbase, wave, lane, ksn, (ks + 1) & 1);

        // ---- A-coefficient build overlaps the async-copy latency
        const float4 xq = *(const float4*)(X + (size_t)r * II + ks * 4);
        const float x0 = halfsel ? xq.y : xq.x;    // feature 4ks + halfsel
        const float x1 = halfsel ? xq.w : xq.z;    // feature 4ks + halfsel + 2
        v16h a;
        build_coeff(a, 0, x0, pts + (ks * 4 + halfsel) * PP);
        build_coeff(a, 8, x1, pts + (ks * 4 + halfsel + 2) * PP);

        // ---- own previous-slab copies done (2 new in flight), then publish
        __builtin_amdgcn_s_wait_asynccnt(2);
        __syncthreads();

        // ---- consume slab from LDS: low-latency ds_load_b128 pairs + 8 WMMAs
        #pragma unroll
        for (int ct = 0; ct < 8; ++ct) {
            const v16h b = *(const v16h*)(&sbuf[q][ct][lane][0]);
            acc[ct] = __builtin_amdgcn_wmma_f32_16x16x32_f16(
                false, a, false, b, (short)0, acc[ct], false, false);
        }
        __syncthreads();                           // buffer q safe to overwrite
    }
#else
    for (int ks = 0; ks < KSLABS; ++ks) {
        v16h b[8];
        #pragma unroll
        for (int ct = 0; ct < 8; ++ct)
            b[ct] = *(const v16h*)(Bp +
                ((((size_t)(colbase + ct) * KSLABS + ks) * 32 + lane) << 4));

        const float4 xq = *(const float4*)(X + (size_t)r * II + ks * 4);
        const float x0 = halfsel ? xq.y : xq.x;
        const float x1 = halfsel ? xq.w : xq.z;
        v16h a;
        build_coeff(a, 0, x0, pts + (ks * 4 + halfsel) * PP);
        build_coeff(a, 8, x1, pts + (ks * 4 + halfsel + 2) * PP);

        #pragma unroll
        for (int ct = 0; ct < 8; ++ct)
            acc[ct] = __builtin_amdgcn_wmma_f32_16x16x32_f16(
                false, a, false, b[ct], (short)0, acc[ct], false, false);
    }
#endif

    // C layout: VGPR j, lanes 0-15 -> M=j; lanes 16-31 -> M=j+8; N = lane&15
    #pragma unroll
    for (int ct = 0; ct < 8; ++ct) {
        const int col = blockIdx.y * 128 + ct * 16 + (lane & 15);
        #pragma unroll
        for (int j = 0; j < 8; ++j) {
            const int rr = row0 + j + halfsel * 8;
            out[(size_t)rr * SS + col] = acc[ct][j];
        }
    }
}

// ---------------------------------------------------------------------------
// Chunked linear-recurrence scan: h[t] = (1-z)h[t-1] + z*hbar = h + z*(hbar-h)
// ---------------------------------------------------------------------------
__device__ __forceinline__ float sigmoidf_(float v) {
    return 1.0f / (1.0f + __expf(-v));
}

// Pass 1: per (b,s,chunk) compute chunk transfer (A = prod a, hpart = result w/ h_in=0)
__global__ __launch_bounds__(256) void scan_pass1(const float* __restrict__ zlin,
                                                  const float* __restrict__ hbar,
                                                  float2* __restrict__ summ) {
    const int tid = blockIdx.x * 256 + threadIdx.x;   // B*S*CHUNKS = 131072
    const int bs  = tid & (BB * SS - 1);
    const int c   = tid / (BB * SS);
    const int b   = bs >> 9, s = bs & (SS - 1);
    const size_t base = ((size_t)(b * TT + c * CLEN)) * SS + s;
    float A = 1.0f, h = 0.0f;
    for (int t = 0; t < CLEN; ++t) {
        const float z  = sigmoidf_(zlin[base + (size_t)t * SS]);
        const float hb = hbar[base + (size_t)t * SS];
        h = (1.0f - z) * h + z * hb;
        A *= (1.0f - z);
    }
    summ[(size_t)c * (BB * SS) + bs] = make_float2(A, h);
}

// Pass 2: sequential combine of 64 chunk summaries per (b,s); records chunk h_in.
__global__ __launch_bounds__(256) void scan_pass2(const float2* __restrict__ summ,
                                                  const float* __restrict__ h0,
                                                  float* __restrict__ carry) {
    const int bs = blockIdx.x * 256 + threadIdx.x;    // B*S = 2048
    const int b  = bs >> 9, s = bs & (SS - 1);
    float h = h0[b * SS + s];
    for (int c = 0; c < CHUNKS; ++c) {
        carry[(size_t)c * (BB * SS) + bs] = h;
        const float2 v = summ[(size_t)c * (BB * SS) + bs];
        h = v.x * h + v.y;
    }
}

// Pass 3: replay each chunk with its true h_in, writing h[t].
__global__ __launch_bounds__(256) void scan_pass3(const float* __restrict__ zlin,
                                                  const float* __restrict__ hbar,
                                                  const float* __restrict__ carry,
                                                  float* __restrict__ out) {
    const int tid = blockIdx.x * 256 + threadIdx.x;   // B*S*CHUNKS
    const int bs  = tid & (BB * SS - 1);
    const int c   = tid / (BB * SS);
    const int b   = bs >> 9, s = bs & (SS - 1);
    const size_t base = ((size_t)(b * TT + c * CLEN)) * SS + s;
    float h = carry[(size_t)c * (BB * SS) + bs];
    for (int t = 0; t < CLEN; ++t) {
        const float z  = sigmoidf_(zlin[base + (size_t)t * SS]);
        const float hb = hbar[base + (size_t)t * SS];
        h = h + z * (hb - h);
        out[base + (size_t)t * SS] = h;
    }
}

// ---------------------------------------------------------------------------
// In-place per-row max-abs normalization over S=512 (1 block per row).
// ---------------------------------------------------------------------------
__global__ __launch_bounds__(256) void rownorm_kernel(float* __restrict__ o) {
    __shared__ float red[8];
    float* p = o + (size_t)blockIdx.x * SS;
    const float v0 = p[threadIdx.x];
    const float v1 = p[threadIdx.x + 256];
    float m = fmaxf(fabsf(v0), fabsf(v1));
    #pragma unroll
    for (int off = 16; off > 0; off >>= 1)
        m = fmaxf(m, __shfl_xor(m, off, 32));
    if ((threadIdx.x & 31) == 0) red[threadIdx.x >> 5] = m;
    __syncthreads();
    float mm = red[0];
    #pragma unroll
    for (int k = 1; k < 8; ++k) mm = fmaxf(mm, red[k]);
    const float inv = 1.0f / (mm + 1e-6f);
    p[threadIdx.x]       = v0 * inv;
    p[threadIdx.x + 256] = v1 * inv;
}

// ---------------------------------------------------------------------------
extern "C" void kernel_launch(void* const* d_in, const int* in_sizes, int n_in,
                              void* d_out, int out_size, void* d_ws, size_t ws_size,
                              hipStream_t stream) {
    (void)in_sizes; (void)n_in; (void)out_size; (void)ws_size;
    const float* x     = (const float*)d_in[0];   // [B,T,I]
    const float* h0    = (const float*)d_in[1];   // [B,S]
    const float* z_pts = (const float*)d_in[2];   // [I,P]
    const float* z_w   = (const float*)d_in[3];   // [I,P,S]
    const float* h_pts = (const float*)d_in[4];   // [I,P]
    const float* h_w   = (const float*)d_in[5];   // [I,P,S]
    float* out = (float*)d_out;                   // [B,T,S]

    char* ws = (char*)d_ws;
    _Float16* bp_h  = (_Float16*)(ws);                        //  4 MB
    _Float16* bp_z  = (_Float16*)(ws + ((size_t)4  << 20));   //  4 MB
    float*    hbar  = (float*)   (ws + ((size_t)8  << 20));   // 16 MB
    float*    zlin  = (float*)   (ws + ((size_t)24 << 20));   // 16 MB
    float2*   summ  = (float2*)  (ws + ((size_t)40 << 20));   //  1 MB
    float*    carry = (float*)   (ws + ((size_t)41 << 20));   // 0.5 MB

    const int packN = CTILES * KSLABS * 32 * 16;              // 2,097,152
    pack_b_kernel<<<packN / 256, 256, 0, stream>>>(h_w, bp_h);
    pack_b_kernel<<<packN / 256, 256, 0, stream>>>(z_w, bp_z);

    dim3 g(NN / 128, SS / 128);                               // 64 x 4
    pl_wmma_gemm<<<g, 256, 0, stream>>>(x, h_pts, bp_h, hbar);
    pl_wmma_gemm<<<g, 256, 0, stream>>>(x, z_pts, bp_z, zlin);

    const int scanN = BB * SS * CHUNKS;                       // 131072
    scan_pass1<<<scanN / 256, 256, 0, stream>>>(zlin, hbar, summ);
    scan_pass2<<<(BB * SS) / 256, 256, 0, stream>>>(summ, h0, carry);
    scan_pass3<<<scanN / 256, 256, 0, stream>>>(zlin, hbar, carry, out);

    rownorm_kernel<<<NN, 256, 0, stream>>>(out);
}
